// MAEModel_4011499454930
// MI455X (gfx1250) — compile-verified, matrix operand order checked by reading, not run on hardware
//
#include <hip/hip_runtime.h>
#include <hip/hip_bf16.h>

typedef __attribute__((ext_vector_type(16))) __bf16 v16bf;
typedef __attribute__((ext_vector_type(8)))  __bf16 v8bf;
typedef __attribute__((ext_vector_type(8)))  float  v8f;
typedef unsigned short us;

#define R_TOTAL 4096   // B*P rows
#define DMODEL  256
#define SEQ     1024
#define NHEAD   8
#define HDIM    32

// ---------------- helpers ----------------

__device__ __forceinline__ us f2bf(float x) {
  unsigned u = __builtin_bit_cast(unsigned, x);
  unsigned r = (u + 0x7FFFu + ((u >> 16) & 1u)) >> 16;
  return (us)r;
}

__device__ __forceinline__ unsigned hashu(unsigned x) {
  x ^= x >> 16; x *= 0x7feb352dU;
  x ^= x >> 15; x *= 0x846ca68bU;
  x ^= x >> 16; return x;
}

// Load one 16-lane-row fragment: per lane two contiguous 16B chunks at
// p and p+16 elements -> 2x global_load_b128. Slot k = (i&7)+8*half+16*(i>>3).
__device__ __forceinline__ v16bf ldpair(const us* __restrict__ p) {
  v8bf c0 = *(const v8bf*)(p);
  v8bf c1 = *(const v8bf*)(p + 16);
  return __builtin_shufflevector(c0, c1, 0, 1, 2, 3, 4, 5, 6, 7,
                                 8, 9, 10, 11, 12, 13, 14, 15);
}

// Fragment where lane -> row (r0 + lane&15), contraction dim contiguous.
// Works for A (row=M) and for transposed-B (row=N) identically.
__device__ __forceinline__ v16bf load_frag_row(const us* __restrict__ base,
                                               int r0, int k0, int ldr, int rMax) {
  int lane = threadIdx.x & 31;
  int half = lane >> 4;
  int r = r0 + (lane & 15);
  if (r >= rMax) r = rMax - 1;
  return ldpair(base + (size_t)r * ldr + k0 + 8 * half);
}

__device__ __forceinline__ v8f wmma_bf16(v16bf a, v16bf b, v8f c) {
  return __builtin_amdgcn_wmma_f32_16x16x32_bf16(false, a, false, b, (short)0, c,
                                                 false, false);
}

// ---------------- generic WMMA GEMM ----------------
// C[M,N] = act(A[M,K]bf16 @ Bt[N,K]bf16^T + biasScale*bias[N]);
// K % 32 == 0, N % 64 == 0. 8 waves/block, each wave owns a 16x64 tile
// (4 accumulators -> 4 independent WMMAs per k-step, 4x A reuse).
// storeMode 1: bf16 store transposed per batch: [B*S,256] -> [B,256,1024].
__global__ void gemm_bf16_wmma(const us* __restrict__ A, const us* __restrict__ Bt,
                               const float* __restrict__ bias, float biasScale,
                               float* __restrict__ outF, us* __restrict__ outB,
                               int M, int N, int K, int doRelu, int storeMode) {
  int wid  = threadIdx.x >> 5;
  int lane = threadIdx.x & 31;
  int tile = blockIdx.x * 8 + wid;
  int tilesN = N >> 6;
  int tilesM = (M + 15) >> 4;
  if (tile >= tilesM * tilesN) return;
  int m0 = (tile / tilesN) << 4;
  int n0 = (tile % tilesN) << 6;
  int lane15 = lane & 15, half = lane >> 4;

  int ma = m0 + lane15; if (ma >= M) ma = M - 1;
  const us* pa = A + (size_t)ma * K + 8 * half;
  const us* pb0 = Bt + (size_t)(n0 + 0  + lane15) * K + 8 * half;
  const us* pb1 = Bt + (size_t)(n0 + 16 + lane15) * K + 8 * half;
  const us* pb2 = Bt + (size_t)(n0 + 32 + lane15) * K + 8 * half;
  const us* pb3 = Bt + (size_t)(n0 + 48 + lane15) * K + 8 * half;

  v8f acc0 = {}, acc1 = {}, acc2 = {}, acc3 = {};
  for (int k0 = 0; k0 < K; k0 += 32) {
    v16bf a  = ldpair(pa  + k0);
    v16bf b0 = ldpair(pb0 + k0);
    v16bf b1 = ldpair(pb1 + k0);
    v16bf b2 = ldpair(pb2 + k0);
    v16bf b3 = ldpair(pb3 + k0);
    if (k0 + 32 < K) {
      __builtin_prefetch(pa  + k0 + 32, 0, 3);
      __builtin_prefetch(pb0 + k0 + 32, 0, 3);
      __builtin_prefetch(pb2 + k0 + 32, 0, 3);
    }
    acc0 = wmma_bf16(a, b0, acc0);
    acc1 = wmma_bf16(a, b1, acc1);
    acc2 = wmma_bf16(a, b2, acc2);
    acc3 = wmma_bf16(a, b3, acc3);
  }

#pragma unroll
  for (int t = 0; t < 4; ++t) {
    v8f acc = (t == 0) ? acc0 : (t == 1) ? acc1 : (t == 2) ? acc2 : acc3;
    int n = n0 + t * 16 + lane15;
    float bv = bias ? biasScale * bias[n] : 0.0f;
#pragma unroll
    for (int j = 0; j < 8; ++j) {
      int m = m0 + j + 8 * half;
      if (m >= M) continue;
      float v = acc[j] + bv;
      if (doRelu) v = fmaxf(v, 0.0f);
      size_t o;
      if (storeMode == 1)
        o = ((size_t)(m >> 10) * 256 + n) * 1024 + (m & 1023);
      else
        o = (size_t)m * N + n;
      if (outF) outF[o] = v;
      if (outB) outB[o] = f2bf(v);
    }
  }
}

// ---------------- flash attention ----------------
// Q,K,O: [B*S, D] bf16; Vt: [B, 256, 1024] bf16 (head-transposed).
// One wave per (b,h, 16-row q tile); online softmax; k-tiles of 32;
// every fragment load is 2x b128 (hd contiguous for Q/K^T, kpos contiguous
// for Vt); P tile re-laid C-frag -> A-frag via 1KB per-wave LDS.
__global__ void flash_attn_kernel(const us* __restrict__ Q, const us* __restrict__ Km,
                                  const us* __restrict__ Vt, us* __restrict__ O) {
  __shared__ us Plds[8][16][32];
  int wid = threadIdx.x >> 5, lane = threadIdx.x & 31, half = lane >> 4;
  int task = blockIdx.x * 8 + wid;           // exactly 2048 tasks over 256 blocks
  int qt = task & 63, h = (task >> 6) & 7, b = task >> 9;
  size_t base = (size_t)b * SEQ * DMODEL + h * HDIM;
  const us* vtb = Vt + ((size_t)b * 256 + h * HDIM) * SEQ;
  int q0 = qt << 4;
  v16bf qf = load_frag_row(Q + base, q0, 0, DMODEL, 1 << 30);
  v8f o0 = {}, o1 = {};
  float mr[8], lr[8];
#pragma unroll
  for (int j = 0; j < 8; ++j) { mr[j] = -1e30f; lr[j] = 0.0f; }
  const float sc = 0.17677669529663687f;     // 1/sqrt(32)
  for (int k0 = 0; k0 < SEQ; k0 += 32) {
    // scores: rows = kpos over K (hd contiguous)
    v16bf bk0 = load_frag_row(Km + base, k0,      0, DMODEL, 1 << 30);
    v16bf bk1 = load_frag_row(Km + base, k0 + 16, 0, DMODEL, 1 << 30);
    v8f z = {};
    v8f s0 = wmma_bf16(qf, bk0, z);
    v8f s1 = wmma_bf16(qf, bk1, z);
#pragma unroll
    for (int j = 0; j < 8; ++j) {
      float a0 = s0[j] * sc, a1 = s1[j] * sc;
      float t = fmaxf(a0, a1);
      t = fmaxf(t, __shfl_xor(t, 1));
      t = fmaxf(t, __shfl_xor(t, 2));
      t = fmaxf(t, __shfl_xor(t, 4));
      t = fmaxf(t, __shfl_xor(t, 8));
      float nm = fmaxf(mr[j], t);
      float alpha = __expf(mr[j] - nm);
      float p0 = __expf(a0 - nm), p1 = __expf(a1 - nm);
      float ps = p0 + p1;
      ps += __shfl_xor(ps, 1);
      ps += __shfl_xor(ps, 2);
      ps += __shfl_xor(ps, 4);
      ps += __shfl_xor(ps, 8);
      lr[j] = lr[j] * alpha + ps;
      mr[j] = nm;
      o0[j] *= alpha; o1[j] *= alpha;
      int row = j + 8 * half, col = lane & 15;
      Plds[wid][row][col]      = f2bf(p0);
      Plds[wid][row][col + 16] = f2bf(p1);
    }
    __syncthreads();
    v16bf pf = ldpair(&Plds[wid][lane & 15][8 * half]);   // 2x ds_load_b128
    __syncthreads();
    // O += P(16x32 kpos) x V(32kpos x 16 hd): rows = hd over Vt (kpos contiguous)
    v16bf bv0 = load_frag_row(vtb, 0,  k0, SEQ, 1 << 30);
    v16bf bv1 = load_frag_row(vtb, 16, k0, SEQ, 1 << 30);
    o0 = wmma_bf16(pf, bv0, o0);
    o1 = wmma_bf16(pf, bv1, o1);
  }
#pragma unroll
  for (int j = 0; j < 8; ++j) {
    int m = q0 + j + 8 * half, n = lane & 15;
    float inv = 1.0f / lr[j];
    O[(size_t)(b * SEQ + m) * DMODEL + h * HDIM + n]      = f2bf(o0[j] * inv);
    O[(size_t)(b * SEQ + m) * DMODEL + h * HDIM + 16 + n] = f2bf(o1[j] * inv);
  }
}

// ---------------- elementwise / small kernels ----------------

// dual mask + patches + valid. One block (64 threads = genes) per spot.
__global__ void mask_patches_kernel(const float* __restrict__ expr,
                                    float* __restrict__ maskOut,
                                    float* __restrict__ patches,
                                    float* __restrict__ valid) {
  int spot = blockIdx.x;            // 0..4095 = b*1024 + h*32 + w
  int c = threadIdx.x;              // 0..63
  int b = spot >> 10, p = spot & 1023;
  float xv = expr[((size_t)(b * 64 + c)) * 1024 + p];
  __shared__ float sx[64];
  __shared__ float su[64];
  unsigned uh = hashu((unsigned)(spot * 64 + c) ^ 0xA511E9B3u);
  su[c] = (float)uh * 2.3283064365386963e-10f;
  sx[c] = xv;
  __syncthreads();
  float s = 0.0f;
  for (int i = 0; i < 64; ++i) s += sx[i];
  bool has = s > 0.0f;
  float mu = su[c];
  int rank = 0;
  for (int i = 0; i < 64; ++i) {
    float o = su[i];
    rank += (o < mu) || (o == mu && i < c);
  }
  unsigned rh = hashu((unsigned)spot ^ 0x9E3779B9u);
  int num = 16 * (int)((rh % 3u) + 1u);           // floor(64 * {.25,.5,.75})
  bool m = (rank < num) && has;
  float mv = m ? 0.0f : xv;
  maskOut[((size_t)(b * 64 + c)) * 1024 + p] = m ? 1.0f : 0.0f;
  patches[(size_t)spot * 64 + c] = mv;
  __syncthreads();
  sx[c] = mv;
  __syncthreads();
  float s2 = 0.0f;
  for (int i = 0; i < 64; ++i) s2 += sx[i];
  if (c == 0) valid[spot] = (s2 != 0.0f) ? 1.0f : 0.0f;
}

// gid_sum[b,e] = sum_c gene_emb[gene_ids[b,c], e]
__global__ void gene_id_sum_kernel(const float* __restrict__ gene_emb,
                                   const int* __restrict__ gene_ids,
                                   float* __restrict__ gid) {
  int b = blockIdx.x, e = threadIdx.x;
  float a = 0.0f;
  for (int c = 0; c < 64; ++c) {
    int g = gene_ids[b * 64 + c];
    a += gene_emb[(size_t)g * 256 + e];
  }
  gid[b * 256 + e] = a;
}

// S[row,e] = sum_c relu(patches[row,c]*w1[e] + b1[e])  -> bf16
__global__ void gene_val_encode_kernel(const float* __restrict__ patches,
                                       const float* __restrict__ w1,
                                       const float* __restrict__ b1,
                                       us* __restrict__ S) {
  int row = blockIdx.x, e = threadIdx.x;
  __shared__ float pr[64];
  if (threadIdx.x < 64) pr[threadIdx.x] = patches[(size_t)row * 64 + threadIdx.x];
  __syncthreads();
  float w = w1[e], bb = b1[e], acc = 0.0f;
#pragma unroll 8
  for (int c = 0; c < 64; ++c) acc += fmaxf(pr[c] * w + bb, 0.0f);
  S[(size_t)row * 256 + e] = f2bf(acc);
}

// x = (xg + 64*b2 + gid[b] + pos[p]) * valid[row]; write f32 + bf16
__global__ void embed_epilogue_kernel(const float* __restrict__ xg,
                                      const float* __restrict__ b2,
                                      const float* __restrict__ gid,
                                      const float* __restrict__ pos,
                                      const float* __restrict__ valid,
                                      float* __restrict__ x, us* __restrict__ xb) {
  int idx = blockIdx.x * 256 + threadIdx.x;
  int row = idx >> 8, e = idx & 255;
  int b = row >> 10, p = row & 1023;
  float v = (xg[idx] + 64.0f * b2[e] + gid[b * 256 + e] + pos[p * 256 + e]) * valid[row];
  x[idx] = v;
  xb[idx] = f2bf(v);
}

// LayerNorm(a + r) * g + be; optional f32/bf16 outputs. One block per row (N=256).
__global__ void ln_residual_kernel(const float* __restrict__ a,
                                   const float* __restrict__ r,
                                   const float* __restrict__ g,
                                   const float* __restrict__ be,
                                   float* __restrict__ outF, us* __restrict__ outB) {
  const int N = 256;
  int row = blockIdx.x, e = threadIdx.x;
  size_t idx = (size_t)row * N + e;
  float v = a[idx] + (r ? r[idx] : 0.0f);
  __shared__ float red[256];
  red[e] = v; __syncthreads();
  for (int s = 128; s > 0; s >>= 1) { if (e < s) red[e] += red[e + s]; __syncthreads(); }
  float mean = red[0] * (1.0f / N);
  __syncthreads();
  float d = v - mean;
  red[e] = d * d; __syncthreads();
  for (int s = 128; s > 0; s >>= 1) { if (e < s) red[e] += red[e + s]; __syncthreads(); }
  float var = red[0] * (1.0f / N);
  float y = d * rsqrtf(var + 1e-5f) * g[e] + be[e];
  if (outF) outF[idx] = y;
  if (outB) outB[idx] = f2bf(y);
}

// f32 -> {f32 copy, bf16 cast}
__global__ void copy_cast_kernel(const float* __restrict__ s, float* __restrict__ dF,
                                 us* __restrict__ dB, int n) {
  int i = blockIdx.x * 256 + threadIdx.x;
  if (i >= n) return;
  float v = s[i];
  if (dF) dF[i] = v;
  if (dB) dB[i] = f2bf(v);
}

// transposed bf16 weight conversion: dst[n*K+k] = bf16(src[k*N+n])
__global__ void convT_kernel(const float* __restrict__ src, us* __restrict__ dst,
                             int K, int N) {
  int i = blockIdx.x * 256 + threadIdx.x;
  if (i >= K * N) return;
  int n = i / K, k = i % K;
  dst[i] = f2bf(src[(size_t)k * N + n]);
}

// src [B*1024, Cc] -> dst [B, Cc, 1024]
__global__ void transpose_bpc_kernel(const float* __restrict__ src,
                                     float* __restrict__ dst, int Cc) {
  int i = blockIdx.x * 256 + threadIdx.x;
  int total = 4 * 1024 * Cc;
  if (i >= total) return;
  int c = i % Cc;
  int p = (i / Cc) % 1024;
  int b = i / (Cc * 1024);
  dst[((size_t)b * Cc + c) * 1024 + p] = src[i];
}

// cls_total[b,e] = cls[e] + gid[b,e] + 64*b2[e] + sum_k 64*relu(b1[k])*W2[k,e]
__global__ void cls_total_kernel(const float* __restrict__ clsv,
                                 const float* __restrict__ gid,
                                 const float* __restrict__ b1,
                                 const float* __restrict__ b2,
                                 const float* __restrict__ W2, us* __restrict__ ct) {
  int b = blockIdx.x, e = threadIdx.x;
  float acc = clsv[e] + gid[b * 256 + e] + 64.0f * b2[e];
  for (int k = 0; k < 256; ++k)
    acc += 64.0f * fmaxf(b1[k], 0.0f) * W2[k * 256 + e];
  ct[b * 256 + e] = f2bf(acc);
}

// ---------------- host orchestration ----------------

extern "C" void kernel_launch(void* const* d_in, const int* in_sizes, int n_in,
                              void* d_out, int out_size, void* d_ws, size_t ws_size,
                              hipStream_t stream) {
  (void)in_sizes; (void)out_size; (void)ws_size;
  if (n_in < 65) return;

  // jax tree-flatten leaf order (dict keys sorted; 'W' < 'b' < 'g' ...)
  const float* expr     = (const float*)d_in[0];
  const int*   gids     = (const int*)  d_in[1];
  const float* W2       = (const float*)d_in[2];
  const float* b1       = (const float*)d_in[3];
  const float* b2       = (const float*)d_in[4];
  const float* clsv     = (const float*)d_in[5];
  const float* enc_W1   = (const float*)d_in[54];
  const float* enc_W2   = (const float*)d_in[55];
  const float* enc_b1   = (const float*)d_in[56];
  const float* enc_b2   = (const float*)d_in[57];
  const float* gene_emb = (const float*)d_in[58];
  const float* mlp2_W1  = (const float*)d_in[59];
  const float* mlp2_W2  = (const float*)d_in[60];
  const float* mlp2_b1  = (const float*)d_in[61];
  const float* mlp2_b2  = (const float*)d_in[62];
  const float* pos      = (const float*)d_in[63];
  const float* w1       = (const float*)d_in[64];

  float* out_recon = (float*)d_out;              // [4,64,32,32]
  float* out_cls   = out_recon + 262144;         // [4,256]
  float* out_mask  = out_cls + 1024;             // [4,64,32,32]
  float* out_encsp = out_mask + 262144;          // [4,256,32,32]

  const int R = R_TOTAL;
  char* ws = (char*)d_ws;
  size_t off = 0;
  auto alloc = [&](size_t bytes) -> void* {
    void* pp = ws + off;
    off = (off + bytes + 255) & ~(size_t)255;
    return pp;
  };

  float* patches = (float*)alloc((size_t)R * 64 * 4);
  float* valid   = (float*)alloc((size_t)R * 4);
  float* gid     = (float*)alloc(4 * 256 * 4);
  us*    Sb      = (us*)   alloc((size_t)R * 256 * 2);
  float* xg      = (float*)alloc((size_t)R * 256 * 4);
  float* x       = (float*)alloc((size_t)R * 256 * 4);
  us*    xb      = (us*)   alloc((size_t)R * 256 * 2);
  float* encf    = (float*)alloc((size_t)R * 256 * 4);
  float* tmp     = (float*)alloc((size_t)R * 256 * 4);
  us*    qb      = (us*)   alloc((size_t)R * 256 * 2);
  us*    kb      = (us*)   alloc((size_t)R * 256 * 2);
  us*    vt      = (us*)   alloc((size_t)R * 256 * 2);   // [B,256,1024]
  us*    ab      = (us*)   alloc((size_t)R * 256 * 2);
  us*    h1      = (us*)   alloc((size_t)R * 1024 * 2);
  us*    r1      = (us*)   alloc((size_t)R * 64 * 2);
  float* r2      = (float*)alloc((size_t)R * 64 * 4);
  us*    ctb     = (us*)   alloc(16 * 256 * 2);
  us*    clshb   = (us*)   alloc(16 * 256 * 2);
  us*    wW2     = (us*)   alloc(65536 * 2);
  us*    wE1     = (us*)   alloc(65536 * 2);
  us*    wE2     = (us*)   alloc(65536 * 2);
  us*    wM1     = (us*)   alloc(16384 * 2);
  us*    wM2     = (us*)   alloc(4096 * 2);
  us*    wQ[3]; us* wK[3]; us* wV[3]; us* wO[3]; us* wF1[3]; us* wF2[3];
  for (int L = 0; L < 3; ++L) {
    wQ[L]  = (us*)alloc(65536 * 2);
    wK[L]  = (us*)alloc(65536 * 2);
    wV[L]  = (us*)alloc(65536 * 2);
    wO[L]  = (us*)alloc(65536 * 2);
    wF1[L] = (us*)alloc(262144 * 2);
    wF2[L] = (us*)alloc(262144 * 2);
  }

  // transposed bf16 weight conversion: dst is [N,K]
  auto convT = [&](const float* s, us* dst, int K, int N) {
    int n = K * N;
    convT_kernel<<<(n + 255) / 256, 256, 0, stream>>>(s, dst, K, N);
  };
  auto gemm = [&](const us* A, const us* Bt, const float* bias, float bscale,
                  float* oF, us* oB, int M, int N, int K, int relu, int mode) {
    int tiles = ((M + 15) / 16) * (N / 64);
    gemm_bf16_wmma<<<(tiles + 7) / 8, 256, 0, stream>>>(A, Bt, bias, bscale, oF, oB,
                                                        M, N, K, relu, mode);
  };

  // one-time (per launch) weight conversions
  convT(W2, wW2, 256, 256);
  convT(enc_W1, wE1, 256, 256);
  convT(enc_W2, wE2, 256, 256);
  convT(mlp2_W1, wM1, 256, 64);
  convT(mlp2_W2, wM2, 64, 64);
  const float *bq[3], *bk_[3], *bv_[3], *bo_[3], *bf1_[3], *bf2_[3];
  const float *g1_[3], *be1_[3], *g2_[3], *be2_[3];
  for (int L = 0; L < 3; ++L) {
    int base = 6 + 16 * L;   // sorted: Wf1 Wf2 Wk Wo Wq Wv be1 be2 bf1 bf2 bk bo bq bv g1 g2
    convT((const float*)d_in[base + 0], wF1[L], 256, 1024);
    convT((const float*)d_in[base + 1], wF2[L], 1024, 256);
    convT((const float*)d_in[base + 2], wK[L], 256, 256);
    convT((const float*)d_in[base + 3], wO[L], 256, 256);
    convT((const float*)d_in[base + 4], wQ[L], 256, 256);
    convT((const float*)d_in[base + 5], wV[L], 256, 256);
    be1_[L] = (const float*)d_in[base + 6];
    be2_[L] = (const float*)d_in[base + 7];
    bf1_[L] = (const float*)d_in[base + 8];
    bf2_[L] = (const float*)d_in[base + 9];
    bk_[L]  = (const float*)d_in[base + 10];
    bo_[L]  = (const float*)d_in[base + 11];
    bq[L]   = (const float*)d_in[base + 12];
    bv_[L]  = (const float*)d_in[base + 13];
    g1_[L]  = (const float*)d_in[base + 14];
    g2_[L]  = (const float*)d_in[base + 15];
  }

  // 1) mask + patches + valid (writes full mask output region)
  mask_patches_kernel<<<R, 64, 0, stream>>>(expr, out_mask, patches, valid);
  // 2) gene id gather-sum, value encoder, embed GEMM + epilogue
  gene_id_sum_kernel<<<4, 256, 0, stream>>>(gene_emb, gids, gid);
  gene_val_encode_kernel<<<R, 256, 0, stream>>>(patches, w1, b1, Sb);
  gemm(Sb, wW2, nullptr, 0.0f, xg, nullptr, R, 256, 256, 0, 0);
  embed_epilogue_kernel<<<R, 256, 0, stream>>>(xg, b2, gid, pos, valid, x, xb);
  // 3) encoder MLP (reuse qb as hidden)
  gemm(xb, wE1, enc_b1, 1.0f, nullptr, qb, R, 256, 256, 1, 0);
  gemm(qb, wE2, enc_b2, 1.0f, encf, nullptr, R, 256, 256, 0, 0);
  copy_cast_kernel<<<(R * 256 + 255) / 256, 256, 0, stream>>>(encf, x, xb, R * 256);
  // 4) decoder: 3 post-norm transformer layers
  for (int L = 0; L < 3; ++L) {
    gemm(xb, wQ[L], bq[L],  1.0f, nullptr, qb, R, 256, 256, 0, 0);
    gemm(xb, wK[L], bk_[L], 1.0f, nullptr, kb, R, 256, 256, 0, 0);
    gemm(xb, wV[L], bv_[L], 1.0f, nullptr, vt, R, 256, 256, 0, 1);  // head-transposed
    flash_attn_kernel<<<256, 256, 0, stream>>>(qb, kb, vt, ab);
    gemm(ab, wO[L], bo_[L], 1.0f, tmp, nullptr, R, 256, 256, 0, 0);
    ln_residual_kernel<<<R, 256, 0, stream>>>(tmp, x, g1_[L], be1_[L], x, xb);
    gemm(xb, wF1[L], bf1_[L], 1.0f, nullptr, h1, R, 1024, 256, 1, 0);
    gemm(h1, wF2[L], bf2_[L], 1.0f, tmp, nullptr, R, 256, 1024, 0, 0);
    ln_residual_kernel<<<R, 256, 0, stream>>>(tmp, x, g2_[L], be2_[L], x, xb);
  }
  // 5) reconstruction head + output transposes
  gemm(xb, wM1, mlp2_b1, 1.0f, nullptr, r1, R, 64, 256, 1, 0);
  gemm(r1, wM2, mlp2_b2, 1.0f, r2, nullptr, R, 64, 64, 1, 0);
  transpose_bpc_kernel<<<(R * 64 + 255) / 256, 256, 0, stream>>>(r2, out_recon, 64);
  transpose_bpc_kernel<<<(R * 256 + 255) / 256, 256, 0, stream>>>(encf, out_encsp, 256);
  // 6) cls path (M=4 GEMMs with clamped loads / guarded stores)
  cls_total_kernel<<<4, 256, 0, stream>>>(clsv, gid, b1, b2, W2, ctb);
  gemm(ctb, wE1, enc_b1, 1.0f, nullptr, clshb, 4, 256, 256, 1, 0);
  gemm(clshb, wE2, enc_b2, 1.0f, out_cls, nullptr, 4, 256, 256, 0, 0);
}
